// DecoderLayer_71038759075876
// MI455X (gfx1250) — matrix operand order repaired and checked by reading, not compiled, not run on hardware
//
#include <hip/hip_runtime.h>
#include <hip/hip_bf16.h>

// ---------------------------------------------------------------------------
// MI455X (gfx1250) decoder layer: rmsnorm -> QKV GEMM (bf16 WMMA) -> qk-norm
// + RoPE -> flash attention (WMMA, 4 waves share K/V staging) -> Wo GEMM +
// residual -> rmsnorm -> SwiGLU MLP (WMMA, fused silu) -> residual.
// GEMM K-loop is double-buffered with GLOBAL_LOAD_ASYNC_TO_LDS_B128 for the
// bf16 activation tile (ASYNCcnt) and register-staged fp32->bf16 weights.
// ---------------------------------------------------------------------------

typedef __attribute__((ext_vector_type(16))) __bf16 v16bf;
typedef __attribute__((ext_vector_type(8)))  __bf16 v8bf;
typedef __attribute__((ext_vector_type(8)))  float  v8f;

#define SEQ  2048
#define DIM  2048
#define NH   16
#define NKV  8
#define HD   128
#define IDIM 8192
#define LPAST 2048
#define TLEN 4096   // LPAST + SEQ

__device__ __forceinline__ v8f wmma_bf16(v16bf a, v16bf b, v8f c) {
    return __builtin_amdgcn_wmma_f32_16x16x32_bf16(
        /*neg_a=*/false, a, /*neg_b=*/false, b,
        /*c_mod=*/(short)0, c, /*reuse_a=*/false, /*reuse_b=*/false);
}

// CDNA5 async memory->LDS copy (16B per lane), tracked with ASYNCcnt.
__device__ __forceinline__ void async_copy16(unsigned lds_off, const void* gptr) {
    unsigned long long ga = (unsigned long long)gptr;
    asm volatile("global_load_async_to_lds_b128 %0, %1, off"
                 :: "v"(lds_off), "v"(ga) : "memory");
}
__device__ __forceinline__ void wait_async0() {
    asm volatile("s_wait_asynccnt 0x0" ::: "memory");
}

// A-matrix (16x32, 16-bit) fragment: lane m = lane&15 holds row m.
// lanes 0-15: K = {0..7, 16..23}; lanes 16-31: K = {8..15, 24..31}.
__device__ __forceinline__ v16bf make_a_frag(const __bf16* row, int lane) {
    int hi8 = (lane & 16) ? 8 : 0;
    v8bf lo = *(const v8bf*)(row + hi8);
    v8bf hi = *(const v8bf*)(row + 16 + hi8);
    v16bf r;
#pragma unroll
    for (int i = 0; i < 8; ++i) { r[i] = lo[i]; r[i + 8] = hi[i]; }
    return r;
}

// B-matrix (32x16, 16-bit) fragment: lane n = lane&15 holds column n.
__device__ __forceinline__ v16bf make_b_frag(const __bf16* col, int lane) {
    int hi8 = (lane & 16) ? 8 : 0;
    v8bf lo = *(const v8bf*)(col + hi8);
    v8bf hi = *(const v8bf*)(col + 16 + hi8);
    v16bf r;
#pragma unroll
    for (int i = 0; i < 8; ++i) { r[i] = lo[i]; r[i + 8] = hi[i]; }
    return r;
}

// ---------------------------------------------------------------------------
// RMSNorm (fp32 in, bf16 out), one block per row.
// ---------------------------------------------------------------------------
__global__ __launch_bounds__(256)
void rmsnorm_kernel(const float* __restrict__ x, const float* __restrict__ w,
                    __bf16* __restrict__ out, int d) {
    int row = blockIdx.x;
    const float* xr = x + (size_t)row * d;
    float ss = 0.f;
    for (int i = threadIdx.x; i < d; i += 256) { float v = xr[i]; ss += v * v; }
    __shared__ float red[256];
    red[threadIdx.x] = ss;
    __syncthreads();
    for (int st = 128; st > 0; st >>= 1) {
        if (threadIdx.x < st) red[threadIdx.x] += red[threadIdx.x + st];
        __syncthreads();
    }
    float inv = rsqrtf(red[0] / (float)d + 1e-6f);
    for (int i = threadIdx.x; i < d; i += 256)
        out[(size_t)row * d + i] = (__bf16)(xr[i] * inv * w[i]);
}

// ---------------------------------------------------------------------------
// GEMM: Y[M,N] = A[M,K](bf16) @ W[N,K]^T(fp32->bf16), fp32 WMMA accumulate.
// Tile 128x64 per 256-thread block (8 waves).  Double-buffered LDS: the A
// tile is fetched with async load-to-LDS (ASYNCcnt); the W tile is fetched to
// registers early, converted, and stored while WMMAs run.
// ---------------------------------------------------------------------------
__global__ __launch_bounds__(256)
void gemm_bf16_kernel(const __bf16* __restrict__ A, const float* __restrict__ W,
                      int M, int N, int K,
                      const float* __restrict__ resid,
                      const __bf16* __restrict__ gateAux,
                      float* __restrict__ outF, __bf16* __restrict__ outB) {
    const int m0 = blockIdx.y * 128;
    const int n0 = blockIdx.x * 64;
    const int tid = threadIdx.x;
    const int lane = tid & 31;
    const int wv = tid >> 5;

    __shared__ __attribute__((aligned(64))) __bf16 As[2][128 * 32];
    __shared__ __attribute__((aligned(64))) __bf16 Ws[2][64 * 32];

    const int ar = tid >> 1, ahalf = tid & 1;      // A staging: 32B/thread
    const int wr = tid >> 2, wq = tid & 3;         // W staging: 8 floats/thread

    v8f zero = {};
    v8f acc[4];
#pragma unroll
    for (int j = 0; j < 4; ++j) acc[j] = zero;

    // ---- prologue: stage tile 0 into buffer 0 ----
    {
        const __bf16* g = A + (size_t)(m0 + ar) * K + ahalf * 16;
        unsigned lds = (unsigned)(unsigned long long)(const void*)
                       (&As[0][ar * 32 + ahalf * 16]);
        async_copy16(lds, g);
        async_copy16(lds + 16, g + 8);

        const float* ws = W + (size_t)(n0 + wr) * K + wq * 8;
        __bf16* wd = &Ws[0][wr * 32 + wq * 8];
#pragma unroll
        for (int i = 0; i < 8; ++i) wd[i] = (__bf16)ws[i];
        wait_async0();
    }
    __syncthreads();

    int buf = 0;
    for (int k0 = 0; k0 < K; k0 += 32, buf ^= 1) {
        const bool more = (k0 + 32) < K;
        float4 wv0, wv1;
        if (more) {
            // issue next A tile async + next W tile global loads (no wait yet)
            const __bf16* g = A + (size_t)(m0 + ar) * K + (k0 + 32) + ahalf * 16;
            unsigned lds = (unsigned)(unsigned long long)(const void*)
                           (&As[buf ^ 1][ar * 32 + ahalf * 16]);
            async_copy16(lds, g);
            async_copy16(lds + 16, g + 8);
            const float* ws = W + (size_t)(n0 + wr) * K + (k0 + 32) + wq * 8;
            wv0 = *(const float4*)(ws);
            wv1 = *(const float4*)(ws + 4);
            if (k0 + 64 < K)
                __builtin_prefetch(W + (size_t)(n0 + wr) * K + k0 + 64, 0, 0);
        }

        // ---- compute current tile ----
        v16bf a = make_a_frag(&As[buf][(16 * wv + (lane & 15)) * 32], lane);
#pragma unroll
        for (int j = 0; j < 4; ++j) {
            v16bf b = make_b_frag(&Ws[buf][(16 * j + (lane & 15)) * 32], lane);
            acc[j] = wmma_bf16(a, b, acc[j]);
        }

        if (more) {
            __bf16* wd = &Ws[buf ^ 1][wr * 32 + wq * 8];
            wd[0] = (__bf16)wv0.x; wd[1] = (__bf16)wv0.y;
            wd[2] = (__bf16)wv0.z; wd[3] = (__bf16)wv0.w;
            wd[4] = (__bf16)wv1.x; wd[5] = (__bf16)wv1.y;
            wd[6] = (__bf16)wv1.z; wd[7] = (__bf16)wv1.w;
            wait_async0();
        }
        __syncthreads();
    }

    // epilogue — C-matrix layout: lane l, vgpr r -> row = (l&16?8:0)+r, col = l&15
    const int rbase = (lane & 16) ? 8 : 0;
    const int col0 = n0 + (lane & 15);
#pragma unroll
    for (int j = 0; j < 4; ++j) {
#pragma unroll
        for (int r = 0; r < 8; ++r) {
            int row = m0 + 16 * wv + rbase + r;
            int col = col0 + 16 * j;
            size_t idx = (size_t)row * N + col;
            float v = acc[j][r];
            if (resid) v += resid[idx];
            if (gateAux) {
                float g = (float)gateAux[idx];
                v = (g / (1.f + __expf(-g))) * v;   // silu(gate) * up
            }
            if (outF) outF[idx] = v;
            else      outB[idx] = (__bf16)v;
        }
    }
}

// ---------------------------------------------------------------------------
// Per-head RMSNorm + RoPE.  raw: [S][nh*128].  Output bf16 (and optional fp32
// present_key) at [(head*outSeqLen + posOff + s)*128 + d].
// ---------------------------------------------------------------------------
__global__ __launch_bounds__(128)
void qk_norm_rope_kernel(const float* __restrict__ raw, const float* __restrict__ wln,
                         const float* __restrict__ cosT, const float* __restrict__ sinT,
                         __bf16* __restrict__ outbf, float* __restrict__ outf,
                         int nh, int outSeqLen, int posOff) {
    int s = blockIdx.x, hh = blockIdx.y, d = threadIdx.x;
    float v = raw[((size_t)s * nh + hh) * HD + d];
    __shared__ float sh[HD];
    __shared__ float nrm[HD];
    sh[d] = v * v;
    __syncthreads();
    for (int st = 64; st > 0; st >>= 1) {
        if (d < st) sh[d] += sh[d + st];
        __syncthreads();
    }
    float inv = rsqrtf(sh[0] / (float)HD + 1e-6f);
    nrm[d] = v * inv * wln[d];
    __syncthreads();
    float rot = (d < 64) ? -nrm[d + 64] : nrm[d - 64];
    float val = nrm[d] * cosT[(size_t)s * HD + d] + rot * sinT[(size_t)s * HD + d];
    size_t o = ((size_t)hh * outSeqLen + posOff + s) * HD + d;
    outbf[o] = (__bf16)val;
    if (outf) outf[o] = val;
}

// past KV [KV][L][HD] -> present [KV][TL][HD] (fp32 out) + bf16 cache copy
__global__ void past_copy_kernel(const float* __restrict__ past,
                                 float* __restrict__ outf, __bf16* __restrict__ outbf) {
    size_t i = (size_t)blockIdx.x * blockDim.x + threadIdx.x;
    if (i >= (size_t)NKV * LPAST * HD) return;
    int d = i % HD;
    size_t r = i / HD;
    int t = r % LPAST;
    int kv = r / LPAST;
    float v = past[i];
    size_t o = ((size_t)kv * TLEN + t) * HD + d;
    outf[o] = v;
    outbf[o] = (__bf16)v;
}

// new V rows [S][KV][HD] -> present_value[kv][L+s][d] + bf16 cache
__global__ void v_new_kernel(const float* __restrict__ vraw,
                             float* __restrict__ outf, __bf16* __restrict__ outbf) {
    size_t i = (size_t)blockIdx.x * blockDim.x + threadIdx.x;
    if (i >= (size_t)SEQ * NKV * HD) return;
    int d = i % HD;
    size_t r = i / HD;
    int kv = r % NKV;
    int s = r / NKV;
    float v = vraw[i];
    size_t o = ((size_t)kv * TLEN + LPAST + s) * HD + d;
    outf[o] = v;
    outbf[o] = (__bf16)v;
}

// ---------------------------------------------------------------------------
// Flash attention: 4 waves per block, one head, 64 query rows (16 per wave).
// K/V chunks (32 keys) staged once in LDS and shared by all 4 waves.
// QK^T and PV via bf16 WMMA; online softmax with 16-lane shfl_xor butterflies.
// ---------------------------------------------------------------------------
__global__ __launch_bounds__(128)
void attn_kernel(const __bf16* __restrict__ qbf, const __bf16* __restrict__ kbf,
                 const __bf16* __restrict__ vbf, __bf16* __restrict__ outbf) {
    const int h = blockIdx.y;
    const int kv = h >> 1;            // H/KV = 2
    const int tid = threadIdx.x;
    const int lane = tid & 31;
    const int wv = tid >> 5;
    const int sblk = blockIdx.x * 64;
    const int sbase = sblk + wv * 16;

    __shared__ __attribute__((aligned(64))) __bf16 Qs[4][16 * HD];
    __shared__ __attribute__((aligned(64))) __bf16 Kc[32 * HD];
    __shared__ __attribute__((aligned(64))) __bf16 Vt[HD * 32];   // transposed
    __shared__ __attribute__((aligned(64))) __bf16 Pt[4][16 * 32];

    // stage Q tile (16 x 128) per wave
    {
        const uint4* src = (const uint4*)(qbf + ((size_t)h * SEQ + sbase) * HD);
        uint4* dst = (uint4*)Qs[wv];
#pragma unroll
        for (int i = 0; i < 8; ++i) dst[i * 32 + lane] = src[i * 32 + lane];
    }
    __syncthreads();

    v16bf qf[4];
#pragma unroll
    for (int c = 0; c < 4; ++c)
        qf[c] = make_a_frag(Qs[wv] + (lane & 15) * HD + 32 * c, lane);

    v8f zero = {};
    v8f o[8];
#pragma unroll
    for (int c = 0; c < 8; ++c) o[c] = zero;
    float mi[8], li[8];
#pragma unroll
    for (int r = 0; r < 8; ++r) { mi[r] = -1e30f; li[r] = 0.f; }

    const float scale = 0.08838834764831845f;   // 1/sqrt(128)
    const int n0i = lane & 15;
    const int rbase = (lane & 16) ? 8 : 0;
    const int tEnd = LPAST + sblk + 64;         // block-wide causal bound

    for (int t = 0; t < tEnd; t += 32) {
        // stage K chunk (32 x 128) row-major: 512 uint4 over 128 threads
        {
            const uint4* src = (const uint4*)(kbf + ((size_t)kv * TLEN + t) * HD);
            uint4* dst = (uint4*)Kc;
#pragma unroll
            for (int i = 0; i < 4; ++i) dst[i * 128 + tid] = src[i * 128 + tid];
        }
        // stage V chunk transposed: Vt[d][kk] = V[t+kk][d]; thread: kk=tid&31, p=tid>>5
        {
            const int kk = tid & 31, p = tid >> 5;
            const uint4* s4 = (const uint4*)(vbf + ((size_t)kv * TLEN + t + kk) * HD) + p * 4;
#pragma unroll
            for (int i = 0; i < 4; ++i) {
                uint4 u = s4[i];
                const __bf16* e = (const __bf16*)&u;
#pragma unroll
                for (int j = 0; j < 8; ++j) Vt[(p * 32 + i * 8 + j) * 32 + kk] = e[j];
            }
        }
        __syncthreads();

        // scores (16 x 32) = Q (16x128) x K^T (128x32), two 16-col tiles
        v8f s0a = zero, s1a = zero;
#pragma unroll
        for (int c = 0; c < 4; ++c) {
            v16bf b0 = make_b_frag(Kc + (lane & 15) * HD + 32 * c, lane);
            s0a = wmma_bf16(qf[c], b0, s0a);
        }
#pragma unroll
        for (int c = 0; c < 4; ++c) {
            v16bf b1 = make_b_frag(Kc + (16 + (lane & 15)) * HD + 32 * c, lane);
            s1a = wmma_bf16(qf[c], b1, s1a);
        }

        // online softmax per row (row reduction = 16-lane half butterflies)
#pragma unroll
        for (int r = 0; r < 8; ++r) {
            int row = rbase + r;
            int limit = LPAST + sbase + row;
            float x0 = s0a[r] * scale; if (t + n0i > limit)      x0 = -1e30f;
            float x1 = s1a[r] * scale; if (t + 16 + n0i > limit) x1 = -1e30f;
            float cm = fmaxf(x0, x1);
            cm = fmaxf(cm, __shfl_xor(cm, 1));
            cm = fmaxf(cm, __shfl_xor(cm, 2));
            cm = fmaxf(cm, __shfl_xor(cm, 4));
            cm = fmaxf(cm, __shfl_xor(cm, 8));
            float nm = fmaxf(mi[r], cm);
            float al = __expf(mi[r] - nm);
            float p0 = __expf(x0 - nm);
            float p1 = __expf(x1 - nm);
            float ps = p0 + p1;
            ps += __shfl_xor(ps, 1);
            ps += __shfl_xor(ps, 2);
            ps += __shfl_xor(ps, 4);
            ps += __shfl_xor(ps, 8);
            li[r] = li[r] * al + ps;
            mi[r] = nm;
#pragma unroll
            for (int c = 0; c < 8; ++c) o[c][r] *= al;
            Pt[wv][row * 32 + n0i]      = (__bf16)p0;
            Pt[wv][row * 32 + 16 + n0i] = (__bf16)p1;
        }
        __syncthreads();

        // O += P (16x32) x V (32x128)
        v16bf pf = make_a_frag(Pt[wv] + (lane & 15) * 32, lane);
#pragma unroll
        for (int c = 0; c < 8; ++c) {
            v16bf vb = make_b_frag(Vt + (16 * c + (lane & 15)) * 32, lane);
            o[c] = wmma_bf16(pf, vb, o[c]);
        }
        __syncthreads();
    }

    // normalize and write bf16 output [S][H*HD]
#pragma unroll
    for (int c = 0; c < 8; ++c) {
#pragma unroll
        for (int r = 0; r < 8; ++r) {
            int row = rbase + r;
            float v = o[c][r] / li[r];
            outbf[(size_t)(sbase + row) * (NH * HD) + h * HD + 16 * c + n0i] = (__bf16)v;
        }
    }
}

// ---------------------------------------------------------------------------
extern "C" void kernel_launch(void* const* d_in, const int* in_sizes, int n_in,
                              void* d_out, int out_size, void* d_ws, size_t ws_size,
                              hipStream_t stream) {
    (void)in_sizes; (void)n_in; (void)out_size; (void)ws_size;
    const float* h_in      = (const float*)d_in[0];
    const float* cosT      = (const float*)d_in[1];
    const float* sinT      = (const float*)d_in[2];
    const float* past_key  = (const float*)d_in[3];
    const float* past_val  = (const float*)d_in[4];
    const float* Wq        = (const float*)d_in[5];
    const float* Wk        = (const float*)d_in[6];
    const float* Wv        = (const float*)d_in[7];
    const float* Wo        = (const float*)d_in[8];
    const float* Wgate     = (const float*)d_in[9];
    const float* Wup       = (const float*)d_in[10];
    const float* Wdown     = (const float*)d_in[11];
    const float* w_in_ln   = (const float*)d_in[12];
    const float* w_q_ln    = (const float*)d_in[13];
    const float* w_k_ln    = (const float*)d_in[14];
    const float* w_post_ln = (const float*)d_in[15];

    float* out_h  = (float*)d_out;                       // [S][D]
    float* out_pk = out_h + (size_t)SEQ * DIM;           // [KV][TL][HD]
    float* out_pv = out_pk + (size_t)NKV * TLEN * HD;    // [KV][TL][HD]

    // workspace carve-up (~168 MB)
    char* w = (char*)d_ws;
    __bf16* xln    = (__bf16*)w; w += (size_t)SEQ * DIM * 2;
    float*  qraw   = (float*)w;  w += (size_t)SEQ * NH  * HD * 4;
    float*  kraw   = (float*)w;  w += (size_t)SEQ * NKV * HD * 4;
    float*  vraw   = (float*)w;  w += (size_t)SEQ * NKV * HD * 4;
    __bf16* qbf    = (__bf16*)w; w += (size_t)NH  * SEQ  * HD * 2;
    __bf16* kbf    = (__bf16*)w; w += (size_t)NKV * TLEN * HD * 2;
    __bf16* vbf    = (__bf16*)w; w += (size_t)NKV * TLEN * HD * 2;
    __bf16* attnbf = (__bf16*)w; w += (size_t)SEQ * NH * HD * 2;
    float*  h2     = (float*)w;  w += (size_t)SEQ * DIM * 4;
    __bf16* x2     = (__bf16*)w; w += (size_t)SEQ * DIM * 2;
    __bf16* gatebf = (__bf16*)w; w += (size_t)SEQ * IDIM * 2;
    __bf16* actbf  = (__bf16*)w; w += (size_t)SEQ * IDIM * 2;

    // 1) input rmsnorm
    rmsnorm_kernel<<<SEQ, 256, 0, stream>>>(h_in, w_in_ln, xln, DIM);

    // 2) QKV projections
    gemm_bf16_kernel<<<dim3(NH * HD / 64, SEQ / 128), 256, 0, stream>>>(
        xln, Wq, SEQ, NH * HD, DIM, nullptr, nullptr, qraw, nullptr);
    gemm_bf16_kernel<<<dim3(NKV * HD / 64, SEQ / 128), 256, 0, stream>>>(
        xln, Wk, SEQ, NKV * HD, DIM, nullptr, nullptr, kraw, nullptr);
    gemm_bf16_kernel<<<dim3(NKV * HD / 64, SEQ / 128), 256, 0, stream>>>(
        xln, Wv, SEQ, NKV * HD, DIM, nullptr, nullptr, vraw, nullptr);

    // 3) q/k head rmsnorm + rope; fill KV caches (fp32 outputs + bf16 copies)
    qk_norm_rope_kernel<<<dim3(SEQ, NH), 128, 0, stream>>>(
        qraw, w_q_ln, cosT, sinT, qbf, nullptr, NH, SEQ, 0);
    qk_norm_rope_kernel<<<dim3(SEQ, NKV), 128, 0, stream>>>(
        kraw, w_k_ln, cosT, sinT, kbf, out_pk, NKV, TLEN, LPAST);
    {
        size_t n = (size_t)NKV * LPAST * HD;
        past_copy_kernel<<<(unsigned)((n + 255) / 256), 256, 0, stream>>>(past_key, out_pk, kbf);
        past_copy_kernel<<<(unsigned)((n + 255) / 256), 256, 0, stream>>>(past_val, out_pv, vbf);
        size_t m = (size_t)SEQ * NKV * HD;
        v_new_kernel<<<(unsigned)((m + 255) / 256), 256, 0, stream>>>(vraw, out_pv, vbf);
    }

    // 4) flash attention (GQA 16q/8kv heads, causal over 4096 keys)
    attn_kernel<<<dim3(SEQ / 64, NH), 128, 0, stream>>>(qbf, kbf, vbf, attnbf);

    // 5) output projection + residual
    gemm_bf16_kernel<<<dim3(DIM / 64, SEQ / 128), 256, 0, stream>>>(
        attnbf, Wo, SEQ, DIM, NH * HD, h_in, nullptr, h2, nullptr);

    // 6) post rmsnorm
    rmsnorm_kernel<<<SEQ, 256, 0, stream>>>(h2, w_post_ln, x2, DIM);

    // 7) SwiGLU MLP: gate, then up with fused silu(gate)*up, then down + residual
    gemm_bf16_kernel<<<dim3(IDIM / 64, SEQ / 128), 256, 0, stream>>>(
        x2, Wgate, SEQ, IDIM, DIM, nullptr, nullptr, nullptr, gatebf);
    gemm_bf16_kernel<<<dim3(IDIM / 64, SEQ / 128), 256, 0, stream>>>(
        x2, Wup, SEQ, IDIM, DIM, nullptr, gatebf, nullptr, actbf);
    gemm_bf16_kernel<<<dim3(DIM / 64, SEQ / 128), 256, 0, stream>>>(
        actbf, Wdown, SEQ, DIM, IDIM, h2, nullptr, out_h, nullptr);
}
